// DglGinPretrainRepr_41214506172627
// MI455X (gfx1250) — matrix operand order repaired and checked by reading, not compiled, not run on hardware
//
#include <hip/hip_runtime.h>
#include <hip/hip_bf16.h>

typedef __attribute__((ext_vector_type(16))) __bf16 v16bf;
typedef __attribute__((ext_vector_type(8)))  float  v8f;

#define NN   50000
#define NE   200000
#define NG   2048
#define EMBD 300
#define NLAY 5
#define MT   3125   // NN/16 real row tiles
#define MTP  3128   // padded to 8*391 so every wave in a block is live
#define KT1  10     // GEMM1: K=300 -> 320 = 10*32
#define NT1  38     // GEMM1: N=600 -> 608 = 38*16 (even -> 19 pair-groups)
#define KT2  19     // GEMM2: K=600 -> 608 = 19*32
#define NT2P 20     // GEMM2: N=300 -> 320 = 20*16 (even -> 10 pair-groups)

// ---------------- utility fills ----------------
__global__ void fill_f32(float* p, long long n) {
  long long t = (long long)blockIdx.x * blockDim.x + threadIdx.x;
  if (t < n) p[t] = 0.0f;
}
__global__ void fill_bf16(__bf16* p, long long n) {
  long long t = (long long)blockIdx.x * blockDim.x + threadIdx.x;
  if (t < n) p[t] = (__bf16)0.0f;
}

// ---------------- h0 = node_emb0[an] + node_emb1[ch] ----------------
__global__ void init_nodes(const int* __restrict__ an, const int* __restrict__ ch,
                           const float* __restrict__ ne0, const float* __restrict__ ne1,
                           float* __restrict__ nf) {
  long long t = (long long)blockIdx.x * blockDim.x + threadIdx.x;
  if (t >= (long long)NN * EMBD) return;
  int n = (int)(t / EMBD), c = (int)(t % EMBD);
  nf[(long long)n * 1800 + c] = ne0[an[n] * EMBD + c] + ne1[ch[n] * EMBD + c];
}

// ---------------- edge message + scatter-add (u_add_e, segment_sum at dst) ----------------
__global__ void edge_scatter(const float* __restrict__ nf, int colbase,
                             const int* __restrict__ src, const int* __restrict__ dst,
                             const int* __restrict__ bt, const int* __restrict__ bd,
                             const float* __restrict__ ee0, const float* __restrict__ ee1,
                             float* __restrict__ agg) {
  long long t = (long long)blockIdx.x * blockDim.x + threadIdx.x;
  if (t >= (long long)NE * 75) return;             // 300 ch = 75 float4 per edge
  int e = (int)(t / 75), c = (int)(t % 75) * 4;
  int s = src[e], d = dst[e];
  float4 f = *(const float4*)(nf + (long long)s * 1800 + colbase + c);
  float4 x = *(const float4*)(ee0 + (long long)bt[e] * EMBD + c);
  float4 y = *(const float4*)(ee1 + (long long)bd[e] * EMBD + c);
  float* ap = agg + (long long)d * EMBD + c;
  atomicAdd(ap + 0, f.x + x.x + y.x);
  atomicAdd(ap + 1, f.y + x.y + y.y);
  atomicAdd(ap + 2, f.z + x.z + y.z);
  atomicAdd(ap + 3, f.w + x.w + y.w);
}

// ---------------- pack f32 row-major [NN x Kreal] -> bf16 A-fragments [MT][KT][32][16] ----------------
// A 16x32 bf16 layout (ISA 7.12.2): lane = m%16 + 16*((kk/8)&1), i = kk%8 + 8*(kk/16)
__global__ void pack_a(const float* __restrict__ srcM, int lda, int Kreal, int KT,
                       __bf16* __restrict__ dstF) {
  long long t = (long long)blockIdx.x * blockDim.x + threadIdx.x;
  long long total = (long long)MT * KT * 512;
  if (t >= total) return;
  int i    = (int)(t & 15);
  int lane = (int)((t >> 4) & 31);
  int kt   = (int)((t >> 9) % KT);
  int mt   = (int)(t / (512LL * KT));
  int half = lane >> 4;
  int m    = mt * 16 + (lane & 15);
  int kk   = (i < 8) ? (half * 8 + i) : (8 + half * 8 + i);  // inverse of layout
  int k    = kt * 32 + kk;
  float v  = (k < Kreal) ? srcM[(long long)m * lda + k] : 0.0f;
  dstF[t] = (__bf16)v;
}

// ---------------- pack f32 W [Kreal x Nreal] -> bf16 B-fragments [NT][KT][32][16] ----------------
// B 32x16 bf16 layout: lane = n%16 + 16*((kk/8)&1), i = kk%8 + 8*(kk/16)
__global__ void pack_w(const float* __restrict__ W, int Kreal, int Nreal, int KT, int NT,
                       __bf16* __restrict__ dstF) {
  long long t = (long long)blockIdx.x * blockDim.x + threadIdx.x;
  if (t >= (long long)NT * KT * 512) return;
  int i    = (int)(t & 15);
  int lane = (int)((t >> 4) & 31);
  int kt   = (int)((t >> 9) % KT);
  int nt   = (int)(t / (512LL * KT));
  int half = lane >> 4;
  int n    = nt * 16 + (lane & 15);
  int k    = kt * 32 + ((i < 8) ? (half * 8 + i) : (8 + half * 8 + i));
  float v  = (k < Kreal && n < Nreal) ? W[(long long)k * Nreal + n] : 0.0f;
  dstF[t] = (__bf16)v;
}

// ---------------- GEMM1: relu(A @ W1 + b1) -> Afrag2 (bf16, packed for GEMM2) ----------------
// grid = (MTP/8, NT1/2); block = 256 (8 waves). Waves of a block share one B pair-panel via LDS.
__global__ void gemm1_wmma(const __bf16* __restrict__ A, const __bf16* __restrict__ Wf,
                           const float* __restrict__ bias, __bf16* __restrict__ A2) {
  __shared__ __bf16 Bs[2 * KT1 * 512];                       // 20 KB
  const int ntg = blockIdx.y;
  // cooperative load of the two B tiles (contiguous in Wf)
  {
    const uint4* srcv = (const uint4*)(Wf + (long long)(2 * ntg) * KT1 * 512);
    uint4* dstv = (uint4*)Bs;
    for (int idx = threadIdx.x; idx < 2 * KT1 * 512 / 8; idx += 256) dstv[idx] = srcv[idx];
  }
  __syncthreads();

  const int mt   = blockIdx.x * 8 + (threadIdx.x >> 5);      // < MTP, A padded
  const int lane = threadIdx.x & 31;
  const v16bf* Ap  = (const v16bf*)(A + (long long)mt * KT1 * 512);
  const v16bf* Bp  = (const v16bf*)Bs;
  v8f c0 = {}, c1 = {};
#pragma unroll 2
  for (int kt = 0; kt < KT1; ++kt) {
    v16bf a  = Ap[kt * 32 + lane];
    v16bf b0 = Bp[kt * 32 + lane];
    v16bf b1 = Bp[(KT1 + kt) * 32 + lane];
    c0 = __builtin_amdgcn_wmma_f32_16x16x32_bf16(false, a, false, b0, (short)0, c0, false, false);
    c1 = __builtin_amdgcn_wmma_f32_16x16x32_bf16(false, a, false, b1, (short)0, c1, false, false);
  }
  const int half = lane >> 4, nl = lane & 15;
#pragma unroll
  for (int j = 0; j < 2; ++j) {
    int n = (2 * ntg + j) * 16 + nl;                         // hidden col = K index of GEMM2
    if (n < 600) {
      float bv = bias[n];
      int kt2 = n >> 5, kk = n & 31;
      int lhi = 16 * ((kk >> 3) & 1);
      int di  = (kk & 7) + 8 * (kk >> 4);
      const v8f& c = j ? c1 : c0;
#pragma unroll
      for (int r = 0; r < 8; ++r) {
        int m = r + 8 * half;                                // row within this 16-row tile
        float v = fmaxf(c[r] + bv, 0.0f);
        A2[(((long long)mt * KT2 + kt2) * 32 + (lhi + m)) * 16 + di] = (__bf16)v;
      }
    }
  }
}

// ---------------- GEMM2: (A2 @ W2 + b2) -> BN -> (relu) -> nf[:, colbase:colbase+300] ----------------
// grid = (MTP/8, NT2P/2); block = 256 (8 waves).
__global__ void gemm2_wmma(const __bf16* __restrict__ A, const __bf16* __restrict__ Wf,
                           const float* __restrict__ bias,
                           const float* __restrict__ gam, const float* __restrict__ bet,
                           const float* __restrict__ mea, const float* __restrict__ var,
                           int doRelu, int colbase, float* __restrict__ nf) {
  __shared__ __bf16 Bs[2 * KT2 * 512];                       // 38 KB
  const int ntg = blockIdx.y;
  {
    const uint4* srcv = (const uint4*)(Wf + (long long)(2 * ntg) * KT2 * 512);
    uint4* dstv = (uint4*)Bs;
    for (int idx = threadIdx.x; idx < 2 * KT2 * 512 / 8; idx += 256) dstv[idx] = srcv[idx];
  }
  __syncthreads();

  const int mt   = blockIdx.x * 8 + (threadIdx.x >> 5);
  const int lane = threadIdx.x & 31;
  const v16bf* Ap = (const v16bf*)(A + (long long)mt * KT2 * 512);
  const v16bf* Bp = (const v16bf*)Bs;
  v8f c0 = {}, c1 = {};
#pragma unroll 2
  for (int kt = 0; kt < KT2; ++kt) {
    v16bf a  = Ap[kt * 32 + lane];
    v16bf b0 = Bp[kt * 32 + lane];
    v16bf b1 = Bp[(KT2 + kt) * 32 + lane];
    c0 = __builtin_amdgcn_wmma_f32_16x16x32_bf16(false, a, false, b0, (short)0, c0, false, false);
    c1 = __builtin_amdgcn_wmma_f32_16x16x32_bf16(false, a, false, b1, (short)0, c1, false, false);
  }
  const int half = lane >> 4, nl = lane & 15;
#pragma unroll
  for (int j = 0; j < 2; ++j) {
    int n = (2 * ntg + j) * 16 + nl;
    if (n < EMBD) {
      float bv = bias[n];
      float sc = rsqrtf(var[n] + 1e-5f) * gam[n];
      float mu = mea[n], be = bet[n];
      const v8f& c = j ? c1 : c0;
#pragma unroll
      for (int r = 0; r < 8; ++r) {
        int m = mt * 16 + r + 8 * half;
        if (m < NN) {
          float v = c[r] + bv;
          v = (v - mu) * sc + be;
          if (doRelu) v = fmaxf(v, 0.0f);
          nf[(long long)m * 1800 + colbase + n] = v;
        }
      }
    }
  }
}

// ---------------- graph pooling ----------------
__global__ void pool_scatter(const float* __restrict__ nf, const int* __restrict__ gid,
                             float* __restrict__ sums) {
  long long t = (long long)blockIdx.x * blockDim.x + threadIdx.x;
  if (t >= (long long)NN * 1800) return;
  int n = (int)(t / 1800), c = (int)(t % 1800);
  atomicAdd(&sums[(long long)gid[n] * 1800 + c], nf[t]);
}
__global__ void pool_count(const int* __restrict__ gid, float* __restrict__ counts) {
  long long t = (long long)blockIdx.x * blockDim.x + threadIdx.x;
  if (t >= NN) return;
  atomicAdd(&counts[gid[t]], 1.0f);
}
__global__ void pool_final(const float* __restrict__ sums, const float* __restrict__ counts,
                           float* __restrict__ out) {
  long long t = (long long)blockIdx.x * blockDim.x + threadIdx.x;
  if (t >= (long long)NG * 1800) return;
  int g = (int)(t / 1800), c = (int)(t % 1800);
  float s = sums[t];
  float cnt = fmaxf(counts[g], 1.0f);
  out[(long long)g * 3600 + c] = s / cnt;
  out[(long long)g * 3600 + 1800 + c] = s;
}

static inline int nblk(long long n) { return (int)((n + 255) / 256); }

extern "C" void kernel_launch(void* const* d_in, const int* in_sizes, int n_in,
                              void* d_out, int out_size, void* d_ws, size_t ws_size,
                              hipStream_t stream) {
  (void)in_sizes; (void)n_in; (void)out_size; (void)ws_size;
  const int*   an  = (const int*)d_in[0];
  const int*   ch  = (const int*)d_in[1];
  const int*   bt  = (const int*)d_in[2];
  const int*   bd  = (const int*)d_in[3];
  const int*   src = (const int*)d_in[4];
  const int*   dst = (const int*)d_in[5];
  const int*   gid = (const int*)d_in[6];
  const float* ne0 = (const float*)d_in[8];
  const float* ne1 = (const float*)d_in[9];
  const float* ee0 = (const float*)d_in[10];
  const float* ee1 = (const float*)d_in[11];
  const float* W1  = (const float*)d_in[12];
  const float* b1  = (const float*)d_in[13];
  const float* W2  = (const float*)d_in[14];
  const float* b2  = (const float*)d_in[15];
  const float* gam = (const float*)d_in[16];
  const float* bet = (const float*)d_in[17];
  const float* mea = (const float*)d_in[18];
  const float* var = (const float*)d_in[19];
  float* out = (float*)d_out;

  // workspace carve-up (every section is a multiple of 256 B -> 32B alignment everywhere)
  float*  nf     = (float*)d_ws;                        // [NN][1800] concat feats
  float*  agg    = nf + (long long)NN * 1800;           // [NN][300]
  float*  sums   = agg + (long long)NN * EMBD;          // [NG][1800]
  float*  counts = sums + (long long)NG * 1800;         // [NG]
  __bf16* A1  = (__bf16*)(counts + NG);                 // [MTP][KT1][32][16]
  __bf16* A2  = A1 + (long long)MTP * KT1 * 512;        // [MTP][KT2][32][16]
  __bf16* W1f = A2 + (long long)MTP * KT2 * 512;        // [NT1][KT1][32][16]
  __bf16* W2f = W1f + (long long)NT1 * KT1 * 512;       // [NT2P][KT2][32][16]

  const int TB = 256;

  // zero accumulators; zero A1/A2 fully once so K-pads and M-pad tiles start clean
  fill_f32 <<<nblk((long long)NG * 1800 + NG), TB, 0, stream>>>(sums, (long long)NG * 1800 + NG);
  fill_bf16<<<nblk((long long)MTP * (KT1 + KT2) * 512), TB, 0, stream>>>(
      A1, (long long)MTP * (KT1 + KT2) * 512);
  init_nodes<<<nblk((long long)NN * EMBD), TB, 0, stream>>>(an, ch, ne0, ne1, nf);

  for (int l = 0; l < NLAY; ++l) {
    fill_f32<<<nblk((long long)NN * EMBD), TB, 0, stream>>>(agg, (long long)NN * EMBD);
    edge_scatter<<<nblk((long long)NE * 75), TB, 0, stream>>>(
        nf, l * EMBD, src, dst, bt, bd,
        ee0 + (long long)l * 6 * EMBD, ee1 + (long long)l * 3 * EMBD, agg);
    pack_a<<<nblk((long long)MT * KT1 * 512), TB, 0, stream>>>(agg, EMBD, EMBD, KT1, A1);
    pack_w<<<nblk((long long)NT1 * KT1 * 512), TB, 0, stream>>>(
        W1 + (long long)l * 300 * 600, 300, 600, KT1, NT1, W1f);
    gemm1_wmma<<<dim3(MTP / 8, NT1 / 2), TB, 0, stream>>>(A1, W1f, b1 + (long long)l * 600, A2);
    pack_w<<<nblk((long long)NT2P * KT2 * 512), TB, 0, stream>>>(
        W2 + (long long)l * 600 * 300, 600, 300, KT2, NT2P, W2f);
    gemm2_wmma<<<dim3(MTP / 8, NT2P / 2), TB, 0, stream>>>(
        A2, W2f, b2 + (long long)l * 300,
        gam + (long long)l * 300, bet + (long long)l * 300,
        mea + (long long)l * 300, var + (long long)l * 300,
        (l < NLAY - 1) ? 1 : 0, (l + 1) * EMBD, nf);
  }

  pool_scatter<<<nblk((long long)NN * 1800), TB, 0, stream>>>(nf, gid, sums);
  pool_count  <<<nblk(NN), TB, 0, stream>>>(gid, counts);
  pool_final  <<<nblk((long long)NG * 1800), TB, 0, stream>>>(sums, counts, out);
}
